// SpectralAggregator_4071628997148
// MI455X (gfx1250) — matrix operand order repaired and checked by reading, not compiled, not run on hardware
//
#include <hip/hip_runtime.h>
#include <math.h>

// ---------------- model dims ----------------
#define DD    256
#define HH    8
#define HD    32
#define LNUM  6
#define NLAT  64
#define ZDIM  128
#define NQS   32
#define BB    64
#define NN    256
#define DFF   1024

typedef __attribute__((ext_vector_type(16))) _Float16 v16h;
typedef __attribute__((ext_vector_type(8)))  _Float16 v8h;
typedef __attribute__((ext_vector_type(8)))  float    v8f;

#define GF_GELU 1
#define GF_RES  2

__device__ inline v16h make_frag(const _Float16* lo, const _Float16* hi) {
    union { v16h v; v8h h[2]; } u;
    u.h[0] = *(const v8h*)lo;
    u.h[1] = *(const v8h*)hi;
    return u.v;
}

__device__ inline v8f vzero8() {
    v8f r;
    #pragma unroll
    for (int i = 0; i < 8; ++i) r[i] = 0.0f;
    return r;
}

__device__ inline float gelu_exact(float x) {
    return 0.5f * x * (1.0f + erff(x * 0.70710678118654752f));
}

// ---------------- GEMM: Y = act(X[MxK] @ W[KxN] + bias) (+res) ----------------
// block 256 threads (8 waves). tile: M=128, N=64, K-step 32.
// LDS double-buffered: one barrier per K-step; global prefetch of tile k+2.
__global__ __launch_bounds__(256)
void k_gemm(const float* __restrict__ X, const float* __restrict__ W,
            const float* __restrict__ bias, const float* __restrict__ res,
            float* __restrict__ Y, int M, int K, int N, int flags)
{
    __shared__ __align__(16) _Float16 As[2][128 * 32];   // [row][k]
    __shared__ __align__(16) _Float16 Bs[2][64 * 32];    // [col][k] (transposed)

    const int tid  = threadIdx.x;
    const int wave = tid >> 5;
    const int lane = tid & 31;
    const int nBase = blockIdx.x * 64;
    const int mBase = blockIdx.y * 128;

    // staging coordinates (fixed per thread)
    const int ar   = tid >> 1;            // A row 0..127
    const int aseg = (tid & 1) * 16;      // A col segment
    int ag = mBase + ar; if (ag >= M) ag = M - 1;
    const int bk   = tid >> 3;            // B k 0..31
    const int bn0  = (tid & 7) * 8;       // B col group

    auto loadA = [&](int kb, int buf) {
        const float* src = X + (size_t)ag * K + (size_t)kb * 32 + aseg;
        _Float16* dst = &As[buf][ar * 32 + aseg];
        #pragma unroll
        for (int i = 0; i < 4; ++i) {
            float4 f = ((const float4*)src)[i];
            dst[i * 4 + 0] = (_Float16)f.x;
            dst[i * 4 + 1] = (_Float16)f.y;
            dst[i * 4 + 2] = (_Float16)f.z;
            dst[i * 4 + 3] = (_Float16)f.w;
        }
    };
    auto loadB = [&](int kb, int buf) {
        const float* src = W + (size_t)(kb * 32 + bk) * N + nBase + bn0;
        #pragma unroll
        for (int i = 0; i < 2; ++i) {
            float4 f = ((const float4*)src)[i];
            Bs[buf][(bn0 + i * 4 + 0) * 32 + bk] = (_Float16)f.x;
            Bs[buf][(bn0 + i * 4 + 1) * 32 + bk] = (_Float16)f.y;
            Bs[buf][(bn0 + i * 4 + 2) * 32 + bk] = (_Float16)f.z;
            Bs[buf][(bn0 + i * 4 + 3) * 32 + bk] = (_Float16)f.w;
        }
    };

    v8f acc[4];
    #pragma unroll
    for (int i = 0; i < 4; ++i) acc[i] = vzero8();

    const int kTiles = K >> 5;
    loadA(0, 0);
    loadB(0, 0);
    __syncthreads();

    for (int kb = 0; kb < kTiles; ++kb) {
        const int cur = kb & 1, nxt = cur ^ 1;
        if (kb + 1 < kTiles) {            // stage next tile into other buffer
            loadA(kb + 1, nxt);
            loadB(kb + 1, nxt);
        }
        if (kb + 2 < kTiles) {            // prefetch tile after next into caches
            __builtin_prefetch(X + (size_t)ag * K + (size_t)(kb + 2) * 32 + aseg, 0, 1);
            __builtin_prefetch(W + (size_t)((kb + 2) * 32 + bk) * N + nBase + bn0, 0, 1);
        }

        // per-wave 16x64 via 4 WMMAs from current buffer
        {
            const int row  = wave * 16 + (lane & 15);
            const int aoff = (lane < 16) ? 0 : 8;
            v16h af = make_frag(&As[cur][row * 32 + aoff],
                                &As[cur][row * 32 + aoff + 16]);
            const int boff = (lane < 16) ? 0 : 8;
            #pragma unroll
            for (int nt = 0; nt < 4; ++nt) {
                int col = nt * 16 + (lane & 15);
                v16h bf = make_frag(&Bs[cur][col * 32 + boff],
                                    &Bs[cur][col * 32 + boff + 16]);
                acc[nt] = __builtin_amdgcn_wmma_f32_16x16x32_f16(
                    false, af, false, bf, (short)0, acc[nt], false, false);
            }
        }
        __syncthreads();
    }

    // epilogue
    {
        const int colL = lane & 15;
        const int mOff = (lane < 16) ? 0 : 8;
        #pragma unroll
        for (int nt = 0; nt < 4; ++nt) {
            int gcol = nBase + nt * 16 + colL;
            float bv = bias[gcol];
            #pragma unroll
            for (int r = 0; r < 8; ++r) {
                int grow = mBase + wave * 16 + mOff + r;
                if (grow < M) {
                    float v = acc[nt][r] + bv;
                    if (flags & GF_GELU) v = gelu_exact(v);
                    if (flags & GF_RES)  v += res[(size_t)grow * N + gcol];
                    Y[(size_t)grow * N + gcol] = v;
                }
            }
        }
    }
}

// ---------------- LayerNorm over D=256, one wave per row ----------------
__global__ __launch_bounds__(256)
void k_layernorm(const float* __restrict__ X, const float* __restrict__ g,
                 const float* __restrict__ b, float* __restrict__ Y, int M)
{
    int wave = threadIdx.x >> 5, lane = threadIdx.x & 31;
    int row = blockIdx.x * 8 + wave;
    if (row >= M) return;
    const float* x = X + (size_t)row * DD;
    float v[8], s = 0.f, s2 = 0.f;
    #pragma unroll
    for (int i = 0; i < 8; ++i) {
        v[i] = x[lane + i * 32]; s += v[i]; s2 += v[i] * v[i];
    }
    #pragma unroll
    for (int off = 16; off >= 1; off >>= 1) {
        s  += __shfl_xor(s,  off, 32);
        s2 += __shfl_xor(s2, off, 32);
    }
    float mean = s * (1.0f / DD);
    float var  = s2 * (1.0f / DD) - mean * mean;
    float inv  = rsqrtf(var + 1e-5f);
    float* y = Y + (size_t)row * DD;
    #pragma unroll
    for (int i = 0; i < 8; ++i) {
        int c = lane + i * 32;
        y[c] = (v[i] - mean) * inv * g[c] + b[c];
    }
}

// ---------------- RoPE on q,k inside qkv (B,N,3,H,HD) ----------------
__global__ void k_rope(float* __restrict__ qkv, const float* __restrict__ wl)
{
    int idx = blockIdx.x * blockDim.x + threadIdx.x;
    if (idx >= BB * NN * HH * 16) return;
    int j = idx & 15;
    int h = (idx >> 4) & 7;
    int t = idx >> 7;                       // token = b*N+n
    float w    = wl[t];
    float invf = __expf(-(float)j * (9.21034037198f / 16.0f)); // 10000^(-j/16)
    float ph   = w * invf;
    float c = __cosf(ph), s = __sinf(ph);
    size_t qb = ((size_t)t * 3 + 0) * DD + h * HD;
    size_t kb = ((size_t)t * 3 + 1) * DD + h * HD;
    float x1, x2;
    x1 = qkv[qb + j]; x2 = qkv[qb + j + 16];
    qkv[qb + j] = x1 * c - x2 * s; qkv[qb + j + 16] = x2 * c + x1 * s;
    x1 = qkv[kb + j]; x2 = qkv[kb + j + 16];
    qkv[kb + j] = x1 * c - x2 * s; qkv[kb + j + 16] = x2 * c + x1 * s;
}

// ---------------- Fused attention: 32 queries x 256 keys, one (b,h) per WG ----------------
union ScoreProb { float s[32 * 256]; _Float16 p[32 * 256]; };

__global__ __launch_bounds__(64)
void k_attn(const float* __restrict__ Q, const float* __restrict__ K,
            const float* __restrict__ V, float* __restrict__ O,
            int qTok, long long qBat, int kvTok, long long kvBat,
            int oTok, long long oBat)
{
    __shared__ __align__(16) _Float16 Qs[32 * 32];       // [q][hd], pre-scaled
    __shared__ __align__(16) _Float16 Ks[2][32 * 32];    // [key][hd] ping-pong
    __shared__ __align__(16) _Float16 Vt[32 * 256];      // [hd][key]
    __shared__ __align__(16) ScoreProb sp;               // scores f32 / probs f16

    const int tid  = threadIdx.x;
    const int wave = tid >> 5;
    const int lane = tid & 31;
    const int qt0  = blockIdx.x * 32;
    const int h    = blockIdx.y;
    const int b    = blockIdx.z;

    const float* Qb = Q + (size_t)b * qBat  + (size_t)h * HD;
    const float* Kb = K + (size_t)b * kvBat + (size_t)h * HD;
    const float* Vb = V + (size_t)b * kvBat + (size_t)h * HD;

    const int lr  = tid >> 1;            // staging row 0..31
    const int lc0 = (tid & 1) * 16;      // staging col segment

    auto loadKV = [&](int kb, int buf) {
        const float* ks = Kb + (size_t)(kb * 32 + lr) * kvTok + lc0;
        const float* vs = Vb + (size_t)(kb * 32 + lr) * kvTok + lc0;
        #pragma unroll
        for (int i = 0; i < 4; ++i) {
            float4 f = ((const float4*)ks)[i];
            Ks[buf][lr * 32 + lc0 + i * 4 + 0] = (_Float16)f.x;
            Ks[buf][lr * 32 + lc0 + i * 4 + 1] = (_Float16)f.y;
            Ks[buf][lr * 32 + lc0 + i * 4 + 2] = (_Float16)f.z;
            Ks[buf][lr * 32 + lc0 + i * 4 + 3] = (_Float16)f.w;
            float4 g = ((const float4*)vs)[i];
            Vt[(lc0 + i * 4 + 0) * 256 + kb * 32 + lr] = (_Float16)g.x;
            Vt[(lc0 + i * 4 + 1) * 256 + kb * 32 + lr] = (_Float16)g.y;
            Vt[(lc0 + i * 4 + 2) * 256 + kb * 32 + lr] = (_Float16)g.z;
            Vt[(lc0 + i * 4 + 3) * 256 + kb * 32 + lr] = (_Float16)g.w;
        }
    };

    // stage Q (scale folded in)
    {
        const float sc = 0.17677669529663688f;   // 1/sqrt(32)
        const float* src = Qb + (size_t)(qt0 + lr) * qTok + lc0;
        #pragma unroll
        for (int i = 0; i < 4; ++i) {
            float4 f = ((const float4*)src)[i];
            Qs[lr * 32 + lc0 + i * 4 + 0] = (_Float16)(f.x * sc);
            Qs[lr * 32 + lc0 + i * 4 + 1] = (_Float16)(f.y * sc);
            Qs[lr * 32 + lc0 + i * 4 + 2] = (_Float16)(f.z * sc);
            Qs[lr * 32 + lc0 + i * 4 + 3] = (_Float16)(f.w * sc);
        }
    }
    loadKV(0, 0);
    __syncthreads();

    // scores: loop over 8 key-blocks of 32, double-buffered K staging
    for (int kb = 0; kb < 8; ++kb) {
        const int cur = kb & 1, nxt = cur ^ 1;
        if (kb + 1 < 8) loadKV(kb + 1, nxt);
        {
            int row  = wave * 16 + (lane & 15);
            int aoff = (lane < 16) ? 0 : 8;
            v16h af = make_frag(Qs + row * 32 + aoff, Qs + row * 32 + aoff + 16);
            int boff = (lane < 16) ? 0 : 8;
            #pragma unroll
            for (int st = 0; st < 2; ++st) {
                int keyl = st * 16 + (lane & 15);
                v16h bf = make_frag(&Ks[cur][keyl * 32 + boff],
                                    &Ks[cur][keyl * 32 + boff + 16]);
                v8f c = vzero8();
                c = __builtin_amdgcn_wmma_f32_16x16x32_f16(
                        false, af, false, bf, (short)0, c, false, false);
                int coll = kb * 32 + st * 16 + (lane & 15);
                int mo   = (lane < 16) ? 0 : 8;
                #pragma unroll
                for (int r = 0; r < 8; ++r)
                    sp.s[(wave * 16 + mo + r) * 256 + coll] = c[r];
            }
        }
        __syncthreads();
    }

    // softmax per row (overwrite score row with f16 probs; unsafe: lane 2t+1 clobbers row t's scores 128..255 early)
    if (tid < 32) {
        const float* srow = sp.s + tid * 256;
        float mx = -1e30f;
        for (int k2 = 0; k2 < 256; ++k2) mx = fmaxf(mx, srow[k2]);
        float sum = 0.f;
        for (int k2 = 0; k2 < 256; ++k2) sum += __expf(srow[k2] - mx);
        float inv = 1.0f / sum;
        _Float16* prow = sp.p + tid * 256;
        for (int k2 = 0; k2 < 256; ++k2)
            prow[k2] = (_Float16)(__expf(srow[k2] - mx) * inv);
    }
    __syncthreads();

    // P @ V : K-dim = 256 keys, N = 32 (hd)
    v8f acc0 = vzero8(), acc1 = vzero8();
    for (int kb = 0; kb < 8; ++kb) {
        int row  = wave * 16 + (lane & 15);
        int aoff = kb * 32 + ((lane < 16) ? 0 : 8);
        v16h af = make_frag(sp.p + row * 256 + aoff, sp.p + row * 256 + aoff + 16);
        int boff = kb * 32 + ((lane < 16) ? 0 : 8);
        int hd0  = lane & 15;
        v16h bf0 = make_frag(Vt + hd0 * 256 + boff, Vt + hd0 * 256 + boff + 16);
        acc0 = __builtin_amdgcn_wmma_f32_16x16x32_f16(
                   false, af, false, bf0, (short)0, acc0, false, false);
        v16h bf1 = make_frag(Vt + (16 + hd0) * 256 + boff, Vt + (16 + hd0) * 256 + boff + 16);
        acc1 = __builtin_amdgcn_wmma_f32_16x16x32_f16(
                   false, af, false, bf1, (short)0, acc1, false, false);
    }

    // store O
    float* Ob = O + (size_t)b * oBat + (size_t)h * HD;
    int colL = lane & 15;
    int mo   = (lane < 16) ? 0 : 8;
    #pragma unroll
    for (int r = 0; r < 8; ++r) {
        int qrow = qt0 + wave * 16 + mo + r;
        Ob[(size_t)qrow * oTok + colL]      = acc0[r];
        Ob[(size_t)qrow * oTok + 16 + colL] = acc1[r];
    }
}

// ---------------- small helpers ----------------
__global__ void k_bcast(const float* __restrict__ src, float* __restrict__ dst,
                        int perBatch, int total)
{
    int i = blockIdx.x * blockDim.x + threadIdx.x;
    if (i >= total) return;
    dst[i] = src[i % perBatch];
}

__global__ void k_mean(const float* __restrict__ det, float* __restrict__ out)
{
    int i = blockIdx.x * blockDim.x + threadIdx.x;   // B*D
    if (i >= BB * DD) return;
    int b = i / DD, d = i % DD;
    float s = 0.f;
    for (int n = 0; n < NLAT; ++n) s += det[((size_t)b * NLAT + n) * DD + d];
    out[i] = s * (1.0f / NLAT);
}

__global__ void k_clip(float* __restrict__ p, int n)
{
    int i = blockIdx.x * blockDim.x + threadIdx.x;
    if (i < n) p[i] = fminf(2.0f, fmaxf(-6.0f, p[i]));
}

// ---------------- host orchestration ----------------
static inline void gemm(const float* X, const float* W, const float* bias,
                        const float* res, float* Y, int M, int K, int N,
                        int flags, hipStream_t st)
{
    dim3 grid(N / 64, (M + 127) / 128);
    k_gemm<<<grid, 256, 0, st>>>(X, W, bias, res, Y, M, K, N, flags);
}

extern "C" void kernel_launch(void* const* d_in, const int* in_sizes, int n_in,
                              void* d_out, int out_size, void* d_ws, size_t ws_size,
                              hipStream_t stream)
{
    const float* h_in = (const float*)d_in[0];
    const float* wl   = (const float*)d_in[1];
    const float* qkv_w = (const float*)d_in[2];
    const float* qkv_b = (const float*)d_in[3];
    const float* out_w = (const float*)d_in[4];
    const float* out_b = (const float*)d_in[5];
    const float* n1_g  = (const float*)d_in[6];
    const float* n1_b  = (const float*)d_in[7];
    const float* ff1_w = (const float*)d_in[8];
    const float* ff1_b = (const float*)d_in[9];
    const float* ff2_w = (const float*)d_in[10];
    const float* ff2_b = (const float*)d_in[11];
    const float* n2_g  = (const float*)d_in[12];
    const float* n2_b  = (const float*)d_in[13];
    const float* latent_q = (const float*)d_in[14];
    const float* dq_w  = (const float*)d_in[15];
    const float* dq_b  = (const float*)d_in[16];
    const float* dkv_w = (const float*)d_in[17];
    const float* dkv_b = (const float*)d_in[18];
    const float* do_w  = (const float*)d_in[19];
    const float* do_b  = (const float*)d_in[20];
    const float* dn_g  = (const float*)d_in[21];
    const float* dn_b  = (const float*)d_in[22];
    const float* dp1_w = (const float*)d_in[23];
    const float* dp1_b = (const float*)d_in[24];
    const float* dp2_w = (const float*)d_in[25];
    const float* dp2_b = (const float*)d_in[26];
    const float* pre1_w = (const float*)d_in[27];
    const float* pre1_b = (const float*)d_in[28];
    const float* pre2_w = (const float*)d_in[29];
    const float* pre2_b = (const float*)d_in[30];
    const float* s_q   = (const float*)d_in[31];
    const float* sq_w  = (const float*)d_in[32];
    const float* sq_b  = (const float*)d_in[33];
    const float* skv_w = (const float*)d_in[34];
    const float* skv_b = (const float*)d_in[35];
    const float* so_w  = (const float*)d_in[36];
    const float* so_b  = (const float*)d_in[37];
    const float* sn_g  = (const float*)d_in[38];
    const float* sn_b  = (const float*)d_in[39];
    const float* mu_w  = (const float*)d_in[40];
    const float* mu_b  = (const float*)d_in[41];
    const float* ls_w  = (const float*)d_in[42];
    const float* ls_b  = (const float*)d_in[43];

    const int M   = BB * NN;                 // 16384 tokens
    float* ws  = (float*)d_ws;
    float* x   = ws;                         // 4,194,304
    float* xn  = x  + 4194304;               // 4,194,304
    float* qkv = xn + 4194304;               // 12,582,912
    float* attno = qkv + 12582912;           // 4,194,304
    float* ffb   = attno + 4194304;          // 16,777,216

    // reuse regions
    float* lqb   = qkv;
    float* detq  = qkv + 1048576;
    float* detkv = qkv + 2097152;
    float* deto  = qkv + 10485760;
    float* det   = qkv + 11534336;
    float* dmean = attno;
    float* tmp1  = attno + 16384;
    float* slq   = attno + 32768;
    float* sqb   = attno + 557056;
    float* sattno= attno + 1081344;
    float* satt  = attno + 1605632;
    float* pret  = ffb;
    float* sbuf  = ffb + 4194304;
    float* skv   = ffb + 8388608;

    float* r_out  = (float*)d_out;           // (64,256)
    float* mu_out = r_out + BB * DD;         // (64,128)
    float* ls_out = mu_out + BB * ZDIM;      // (64,128)

    hipMemcpyAsync(x, h_in, (size_t)M * DD * sizeof(float),
                   hipMemcpyDeviceToDevice, stream);

    // -------- transformer layers --------
    for (int i = 0; i < LNUM; ++i) {
        k_layernorm<<<M / 8, 256, 0, stream>>>(x, n1_g + i * DD, n1_b + i * DD, xn, M);
        gemm(xn, qkv_w + (size_t)i * DD * 3 * DD, qkv_b + (size_t)i * 3 * DD,
             nullptr, qkv, M, DD, 3 * DD, 0, stream);
        k_rope<<<(BB * NN * HH * 16) / 256, 256, 0, stream>>>(qkv, wl);
        {
            dim3 g(NN / 32, HH, BB);
            k_attn<<<g, 64, 0, stream>>>(qkv, qkv + DD, qkv + 2 * DD, attno,
                                         3 * DD, (long long)NN * 3 * DD,
                                         3 * DD, (long long)NN * 3 * DD,
                                         DD, (long long)NN * DD);
        }
        gemm(attno, out_w + (size_t)i * DD * DD, out_b + (size_t)i * DD,
             x, x, M, DD, DD, GF_RES, stream);
        k_layernorm<<<M / 8, 256, 0, stream>>>(x, n2_g + i * DD, n2_b + i * DD, xn, M);
        gemm(xn, ff1_w + (size_t)i * DD * DFF, ff1_b + (size_t)i * DFF,
             nullptr, ffb, M, DD, DFF, GF_GELU, stream);
        gemm(ffb, ff2_w + (size_t)i * DFF * DD, ff2_b + (size_t)i * DD,
             x, x, M, DFF, DD, GF_RES, stream);
    }

    // -------- detector cross-attention path --------
    k_bcast<<<(BB * NLAT * DD) / 256, 256, 0, stream>>>(latent_q, lqb,
                                                        NLAT * DD, BB * NLAT * DD);
    gemm(lqb, dq_w, dq_b, nullptr, detq, BB * NLAT, DD, DD, 0, stream);
    gemm(x, dkv_w, dkv_b, nullptr, detkv, M, DD, 2 * DD, 0, stream);
    {
        dim3 g(NLAT / 32, HH, BB);
        k_attn<<<g, 64, 0, stream>>>(detq, detkv, detkv + DD, deto,
                                     DD, (long long)NLAT * DD,
                                     2 * DD, (long long)NN * 2 * DD,
                                     DD, (long long)NLAT * DD);
    }
    gemm(deto, do_w, do_b, nullptr, det, BB * NLAT, DD, DD, 0, stream);
    k_layernorm<<<(BB * NLAT) / 8, 256, 0, stream>>>(det, dn_g, dn_b, det, BB * NLAT);
    k_mean<<<(BB * DD + 255) / 256, 256, 0, stream>>>(det, dmean);
    gemm(dmean, dp1_w, dp1_b, nullptr, tmp1, BB, DD, DD, GF_GELU, stream);
    gemm(tmp1, dp2_w, dp2_b, nullptr, r_out, BB, DD, DD, 0, stream);

    // -------- spectral query path --------
    gemm(x, pre1_w, pre1_b, nullptr, pret, M, DD, DD, GF_GELU, stream);
    gemm(pret, pre2_w, pre2_b, nullptr, sbuf, M, DD, DD, 0, stream);
    k_bcast<<<(BB * NQS * DD) / 256, 256, 0, stream>>>(s_q, slq,
                                                       NQS * DD, BB * NQS * DD);
    gemm(slq, sq_w, sq_b, nullptr, sqb, BB * NQS, DD, DD, 0, stream);
    gemm(sbuf, skv_w, skv_b, nullptr, skv, M, DD, 2 * DD, 0, stream);
    {
        dim3 g(NQS / 32, HH, BB);
        k_attn<<<g, 64, 0, stream>>>(sqb, skv, skv + DD, sattno,
                                     DD, (long long)NQS * DD,
                                     2 * DD, (long long)NN * 2 * DD,
                                     DD, (long long)NQS * DD);
    }
    gemm(sattno, so_w, so_b, nullptr, satt, BB * NQS, DD, DD, 0, stream);
    k_layernorm<<<(BB * NQS) / 8, 256, 0, stream>>>(satt, sn_g, sn_b, satt, BB * NQS);

    // heads on flattened (B, NQ*D)
    gemm(satt, mu_w, mu_b, nullptr, mu_out, BB, NQS * DD, ZDIM, 0, stream);
    gemm(satt, ls_w, ls_b, nullptr, ls_out, BB, NQS * DD, ZDIM, 0, stream);
    k_clip<<<(BB * ZDIM + 255) / 256, 256, 0, stream>>>(ls_out, BB * ZDIM);
}
